// MixedAttentionLayer_40613210751280
// MI455X (gfx1250) — matrix operand order repaired and checked by reading, not compiled, not run on hardware
//
#include <hip/hip_runtime.h>
#include <hip/hip_bf16.h>

// ---------------------------------------------------------------------------
// MixedAttentionLayer for MI455X (gfx1250, wave32).
// GEMMs: v_wmma_f32_16x16x32_f16, f16 operand mirrors, TDM tensor_load_to_lds
// staging for non-transposed A / transposed B tiles, 16B LDS fragment loads.
// ---------------------------------------------------------------------------

typedef __attribute__((ext_vector_type(16))) _Float16 v16h;
typedef __attribute__((ext_vector_type(8)))  _Float16 v8h;
typedef __attribute__((ext_vector_type(8)))  float    v8f;
typedef __attribute__((ext_vector_type(4)))  unsigned int v4u;
typedef __attribute__((ext_vector_type(8)))  int      v8i;
typedef __attribute__((ext_vector_type(4)))  int      v4i;

#define DIMC 256
#define NC   4096
#define EC   1024
#define INV_SQRT_D 0.0625f
#define NEGBIG -1.0e9f

#define TILE_M 128
#define TILE_N 128
#define TILE_K 32
#define LP     40      // LDS row pitch in halves: 32 data + 4-DWORD pad (80B, 16B aligned)

#if __has_builtin(__builtin_amdgcn_tensor_load_to_lds)
#define HAVE_TDM 1
#endif

// ---------------------------------------------------------------------------
// TDM: DMA a 2D f16 tile (tileW contiguous elems x tileH rows, row stride
// strideElems) from global memory into LDS at byte offset ldsOff, inserting a
// 4-DWORD pad after every 16 DWORDs (one 32-half row) -> LDS pitch = LP halves.
// D# layout per CDNA5 ISA.
// ---------------------------------------------------------------------------
#ifdef HAVE_TDM
__device__ __forceinline__ void tdm_load_2d(const _Float16* gptr, unsigned int ldsOff,
                                            unsigned int tileW, unsigned int tileH,
                                            unsigned int strideElems)
{
    unsigned long long ga = (unsigned long long)(size_t)gptr;
    v4u g0;
    g0[0] = 1u;                                   // count=1, user descriptor
    g0[1] = ldsOff;                               // lds byte address
    g0[2] = (unsigned int)(ga & 0xffffffffu);
    g0[3] = (unsigned int)((ga >> 32) & 0x1ffffffu) | (2u << 30);  // type=2 (image)

    const unsigned int td0 = tileW;               // tensor dims == tile dims: fully in bounds
    const unsigned int td1 = tileH;
    v8i g1;
    g1[0] = (int)((1u << 16)                      // data_size = 2 bytes
                | (1u << 20)                      // pad_enable
                | (3u << 22)                      // pad_interval: 16 DWORDs
                | (3u << 25));                    // pad_amount:   4 DWORDs
    g1[1] = (int)((td0 & 0xffffu) << 16);                         // tensor_dim0[15:0]
    g1[2] = (int)((td0 >> 16) | ((td1 & 0xffffu) << 16));         // td0 hi | tensor_dim1 lo
    g1[3] = (int)((td1 >> 16) | (tileW << 16));                   // td1 hi | tile_dim0
    g1[4] = (int)(tileH & 0xffffu);                               // tile_dim1 (tile_dim2=0)
    g1[5] = (int)strideElems;                                     // tensor_dim0_stride lo
    g1[6] = 0;                                                    // stride hi | dim1_stride lo
    g1[7] = 0;
    v4i z4 = {0, 0, 0, 0};
#if defined(__clang_major__) && __clang_major__ >= 23
    v8i z8 = {0, 0, 0, 0, 0, 0, 0, 0};
    __builtin_amdgcn_tensor_load_to_lds(g0, g1, z4, z4, z8, 0);
#else
    __builtin_amdgcn_tensor_load_to_lds(g0, g1, z4, z4, 0);
#endif
}
#endif

// ---------------------------------------------------------------------------
// Generic WMMA GEMM on f16 operands: C = epi( scale * op(A)[M,K] @ op(B)[K,N] )
// Block 256 threads (8 waves), tile 128x128, wave tile 32x64 (2x4 WMMA).
// A staged as lsA[r][k]; B staged transposed as lsBT[n][k] so both fragment
// reads are contiguous 16B chunks. All M,N %128==0, K %32==0 here.
// ---------------------------------------------------------------------------
__global__ __launch_bounds__(256)
void gemm_wmma_f16(const _Float16* __restrict__ A, int lda, int ta,
                   const _Float16* __restrict__ B, int ldb, int tb,
                   float* __restrict__ C, int ldc, _Float16* __restrict__ C16,
                   const float* __restrict__ Cadd, const float* __restrict__ colDiv,
                   float scale, int M, int N, int K, int do_elu)
{
    __shared__ __align__(16) _Float16 lsA[TILE_M][LP];
    __shared__ __align__(16) _Float16 lsBT[TILE_N][LP];

    const int tid  = threadIdx.x;
    const int lane = tid & 31;
    const int wid  = tid >> 5;
    const int wrow = (wid >> 1) * 32;
    const int wcol = (wid & 1)  * 64;
    const int row0 = blockIdx.y * TILE_M;
    const int col0 = blockIdx.x * TILE_N;
    const int l    = lane & 15;
    const int hi   = lane >> 4;

    v8f acc[2][4];
#pragma unroll
    for (int mi = 0; mi < 2; ++mi)
#pragma unroll
        for (int ni = 0; ni < 4; ++ni)
            acc[mi][ni] = (v8f){0.f,0.f,0.f,0.f,0.f,0.f,0.f,0.f};

    for (int k0 = 0; k0 < K; k0 += TILE_K) {
        // ---------------- stage A tile -> lsA[r][k] ----------------
        if (!ta) {
#ifdef HAVE_TDM
            if (wid == 0)
                tdm_load_2d(A + (size_t)row0 * lda + k0,
                            (unsigned int)(size_t)&lsA[0][0], TILE_K, TILE_M, lda);
#else
            {   // wide manual copy: thread t -> row t>>1, half-row (t&1)*16
                const int r = tid >> 1, ko = (tid & 1) * 16;
                const _Float16* g = A + (size_t)(row0 + r) * lda + k0 + ko;
                *(v8h*)&lsA[r][ko]     = *(const v8h*)g;
                *(v8h*)&lsA[r][ko + 8] = *(const v8h*)(g + 8);
            }
#endif
        } else {
            // memory is [K x M]: coalesced reads along r, scatter b16 to LDS
            const int kk = tid >> 3, rr = (tid & 7) * 16;
            const _Float16* g = A + (size_t)(k0 + kk) * lda + row0 + rr;
            const v8h a0 = *(const v8h*)g;
            const v8h a1 = *(const v8h*)(g + 8);
#pragma unroll
            for (int j = 0; j < 8; ++j) {
                lsA[rr + j][kk]     = a0[j];
                lsA[rr + 8 + j][kk] = a1[j];
            }
        }
        // ---------------- stage B tile -> lsBT[n][k] ----------------
        if (tb) {
            // memory is [N x K]: rows are n with contiguous k -> direct DMA
#ifdef HAVE_TDM
            if (wid == 1)
                tdm_load_2d(B + (size_t)col0 * ldb + k0,
                            (unsigned int)(size_t)&lsBT[0][0], TILE_K, TILE_N, ldb);
#else
            {
                const int n = tid >> 1, ko = (tid & 1) * 16;
                const _Float16* g = B + (size_t)(col0 + n) * ldb + k0 + ko;
                *(v8h*)&lsBT[n][ko]     = *(const v8h*)g;
                *(v8h*)&lsBT[n][ko + 8] = *(const v8h*)(g + 8);
            }
#endif
        } else {
            // memory is [K x N]: coalesced reads along n, scatter b16 to LDS
            const int kk = tid >> 3, nn = (tid & 7) * 16;
            const _Float16* g = B + (size_t)(k0 + kk) * ldb + col0 + nn;
            const v8h b0 = *(const v8h*)g;
            const v8h b1 = *(const v8h*)(g + 8);
#pragma unroll
            for (int j = 0; j < 8; ++j) {
                lsBT[nn + j][kk]     = b0[j];
                lsBT[nn + 8 + j][kk] = b1[j];
            }
        }
#if defined(HAVE_TDM) && __has_builtin(__builtin_amdgcn_s_wait_tensorcnt)
        if (wid < 2) __builtin_amdgcn_s_wait_tensorcnt(0);
#endif
        __syncthreads();

        // ---------------- fragments (ISA 7.12.2 16-bit layouts) ----------------
        v16h afrag[2], bfrag[4];
        const int klo = hi * 8;
        const int kb  = hi * 16;
#pragma unroll
        for (int mi = 0; mi < 2; ++mi) {
            const _Float16* p = &lsA[wrow + mi * 16 + l][klo];
            const v8h lo  = *(const v8h*)p;          // K = klo .. klo+7
            const v8h hi8 = *(const v8h*)(p + 16);   // K = 16+klo .. 16+klo+7
            afrag[mi] = __builtin_shufflevector(lo, hi8,
                0,1,2,3,4,5,6,7,8,9,10,11,12,13,14,15);
        }
#pragma unroll
        for (int ni = 0; ni < 4; ++ni) {
            const _Float16* p = &lsBT[wcol + ni * 16 + l][klo];
            const v8h b0 = *(const v8h*)p;           // K = klo .. klo+7
            const v8h b1 = *(const v8h*)(p + 16);    // K = 16+klo .. 16+klo+7
            bfrag[ni] = __builtin_shufflevector(b0, b1,
                0,1,2,3,4,5,6,7,8,9,10,11,12,13,14,15);
        }

#pragma unroll
        for (int mi = 0; mi < 2; ++mi)
#pragma unroll
            for (int ni = 0; ni < 4; ++ni)
                acc[mi][ni] = __builtin_amdgcn_wmma_f32_16x16x32_f16(
                    false, afrag[mi], false, bfrag[ni],
                    (short)0, acc[mi][ni], false, false);

        __syncthreads();
    }

    // ---------------- epilogue ----------------
#pragma unroll
    for (int mi = 0; mi < 2; ++mi) {
#pragma unroll
        for (int ni = 0; ni < 4; ++ni) {
            const int c     = col0 + wcol + ni * 16 + l;
            const int rbase = row0 + wrow + mi * 16 + hi * 8;
            const float cd  = colDiv ? colDiv[c] : 1.f;
#pragma unroll
            for (int j = 0; j < 8; ++j) {
                const int r = rbase + j;
                float v = acc[mi][ni][j] * scale;
                if (colDiv) v /= cd;
                if (Cadd)   v += Cadd[(size_t)r * ldc + c];
                if (do_elu) v = v > 0.f ? v : (__expf(v) - 1.f);
                if (C)   C  [(size_t)r * ldc + c] = v;
                if (C16) C16[(size_t)r * ldc + c] = (_Float16)v;
            }
        }
    }
}

// ---------------------------------------------------------------------------
__global__ __launch_bounds__(256)
void cvt_f16_kernel(const float* __restrict__ in, _Float16* __restrict__ out, size_t n)
{
    const size_t idx = (size_t)blockIdx.x * 256 + threadIdx.x;
    if (idx < n) out[idx] = (_Float16)in[idx];
}

// sc = (h_he @ u) * inv ; s1 = h_g @ a_g[:D] ; s2 = h_g @ a_g[D:]
__global__ __launch_bounds__(256)
void dots_kernel(const float* __restrict__ hhe, const float* __restrict__ hg,
                 const float* __restrict__ u, const float* __restrict__ ag,
                 float* __restrict__ sc, float* __restrict__ s1, float* __restrict__ s2,
                 float inv)
{
    __shared__ float red[256];
    const int i = blockIdx.x;
    const int t = threadIdx.x;
    const float he  = hhe[(size_t)i * DIMC + t];
    const float hgv = hg[(size_t)i * DIMC + t];

    red[t] = he * u[t]; __syncthreads();
    for (int s = 128; s > 0; s >>= 1) { if (t < s) red[t] += red[t + s]; __syncthreads(); }
    if (t == 0) sc[i] = red[0] * inv;
    __syncthreads();

    red[t] = hgv * ag[t]; __syncthreads();
    for (int s = 128; s > 0; s >>= 1) { if (t < s) red[t] += red[t + s]; __syncthreads(); }
    if (t == 0) s1[i] = red[0];
    __syncthreads();

    red[t] = hgv * ag[DIMC + t]; __syncthreads();
    for (int s = 128; s > 0; s >>= 1) { if (t < s) red[t] += red[t + s]; __syncthreads(); }
    if (t == 0) s2[i] = red[0];
}

// per-hyperedge online max/sum of masked sc, plus hyperedge size
__global__ __launch_bounds__(256)
void col_stats_kernel(const float* __restrict__ H, const float* __restrict__ sc,
                      float* __restrict__ colmax, float* __restrict__ colsum,
                      float* __restrict__ hesz)
{
    const int c = blockIdx.x * 256 + threadIdx.x;
    if (c >= EC) return;
    float m = -1e30f, s = 0.f; int cnt = 0;
    for (int i = 0; i < NC; ++i) {
        if (H[(size_t)i * EC + c] > 0.f) {
            const float v = sc[i];
            if (v > m) { s = s * __expf(m - v) + 1.f; m = v; }
            else       { s += __expf(v - m); }
            ++cnt;
        }
    }
    colmax[c] = m;
    colsum[c] = s;
    hesz[c]   = (float)(cnt < 1 ? 1 : cnt);
}

// A[i,e] = mask ? exp(sc[i]-colmax)/colsum : 0   (f32 + f16 mirror)
__global__ __launch_bounds__(256)
void build_A_kernel(const float* __restrict__ H, const float* __restrict__ sc,
                    const float* __restrict__ colmax, const float* __restrict__ colsum,
                    float* __restrict__ A, _Float16* __restrict__ A16)
{
    const size_t idx = (size_t)blockIdx.x * 256 + threadIdx.x;
    const int i = (int)(idx / EC), c = (int)(idx % EC);
    float a = 0.f;
    if (H[idx] > 0.f) {
        const float s = colsum[c];
        a = s > 0.f ? __expf(sc[i] - colmax[c]) / s : 0.f;
    }
    A[idx]   = a;
    A16[idx] = (_Float16)a;
}

// in-place row softmax; optional NEG pre-mask / zero post-mask; optional f16 out
__global__ __launch_bounds__(256)
void row_softmax_kernel(float* __restrict__ S, const float* __restrict__ mask,
                        int L, int premask, int postmask, _Float16* __restrict__ out16)
{
    __shared__ float red[256];
    const int r = blockIdx.x, t = threadIdx.x;
    const size_t base = (size_t)r * L;

    float m = -1e30f;
    for (int j = t; j < L; j += 256) {
        float v = S[base + j];
        if (premask && mask[base + j] <= 0.f) v = NEGBIG;
        m = fmaxf(m, v);
    }
    red[t] = m; __syncthreads();
    for (int s = 128; s > 0; s >>= 1) { if (t < s) red[t] = fmaxf(red[t], red[t + s]); __syncthreads(); }
    m = red[0]; __syncthreads();

    float sum = 0.f;
    for (int j = t; j < L; j += 256) {
        float v = S[base + j];
        if (premask && mask[base + j] <= 0.f) v = NEGBIG;
        sum += __expf(v - m);
    }
    red[t] = sum; __syncthreads();
    for (int s = 128; s > 0; s >>= 1) { if (t < s) red[t] += red[t + s]; __syncthreads(); }
    sum = red[0];
    const float invs = sum > 0.f ? 1.f / sum : 0.f;

    for (int j = t; j < L; j += 256) {
        float v = S[base + j];
        if (premask && mask[base + j] <= 0.f) v = NEGBIG;
        float o = __expf(v - m) * invs;
        if (postmask && mask[base + j] <= 0.f) o = 0.f;
        S[base + j] = o;
        if (out16) out16[base + j] = (_Float16)o;
    }
}

// masked leaky-relu graph scores (pre-softmax), per reference semantics
__global__ __launch_bounds__(256)
void build_G_kernel(const float* __restrict__ adj, const float* __restrict__ s1,
                    const float* __restrict__ s2, float* __restrict__ G)
{
    const int i = blockIdx.x, t = threadIdx.x;
    const float a = s1[i];
    const size_t base = (size_t)i * NC;
    for (int j = t; j < NC; j += 256) {
        float v = a + s2[j];
        v = v > 0.f ? v : 0.2f * v;
        const float msk = adj[base + j] + (j == i ? 1.f : 0.f);
        G[base + j] = v * msk - 1e9f * fmaxf(1.f - msk, 0.f);
    }
}

// Out[i,e] = column_softmax_i(X+Y)[i,e] * (H>0)   (A_hat stored transposed [N,E])
__global__ __launch_bounds__(256)
void col_softmax_kernel(const float* __restrict__ X, const float* __restrict__ Y,
                        const float* __restrict__ H, float* __restrict__ Out,
                        _Float16* __restrict__ Out16)
{
    const int c = blockIdx.x * 256 + threadIdx.x;
    if (c >= EC) return;
    float m = -1e30f, s = 0.f;
    for (int i = 0; i < NC; ++i) {
        const size_t idx = (size_t)i * EC + c;
        const float v = X[idx] + Y[idx];
        if (v > m) { s = s * __expf(m - v) + 1.f; m = v; }
        else       { s += __expf(v - m); }
    }
    const float invs = s > 0.f ? 1.f / s : 0.f;
    for (int i = 0; i < NC; ++i) {
        const size_t idx = (size_t)i * EC + c;
        float o = __expf(X[idx] + Y[idx] - m) * invs;
        o = H[idx] > 0.f ? o : 0.f;
        Out[idx]   = o;
        Out16[idx] = (_Float16)o;
    }
}

__global__ __launch_bounds__(256)
void add_kernel(const float* __restrict__ X, const float* __restrict__ Y,
                float* __restrict__ Out, size_t count)
{
    const size_t idx = (size_t)blockIdx.x * 256 + threadIdx.x;
    if (idx < count) Out[idx] = X[idx] + Y[idx];
}

// ---------------------------------------------------------------------------
extern "C" void kernel_launch(void* const* d_in, const int* in_sizes, int n_in,
                              void* d_out, int out_size, void* d_ws, size_t ws_size,
                              hipStream_t stream) {
    (void)in_sizes; (void)n_in; (void)out_size; (void)ws_size;

    const float* x    = (const float*)d_in[0];
    const float* H    = (const float*)d_in[1];
    const float* adj  = (const float*)d_in[2];
    const float* W_he = (const float*)d_in[3];
    const float* u    = (const float*)d_in[4];
    const float* W_e  = (const float*)d_in[5];
    const float* W_n  = (const float*)d_in[6];
    const float* W_g  = (const float*)d_in[7];
    const float* a_g  = (const float*)d_in[8];
    float* out = (float*)d_out;

    const size_t ND = (size_t)NC * DIMC;      // 1,048,576
    const size_t ED = (size_t)EC * DIMC;      //   262,144
    const size_t DD = (size_t)DIMC * DIMC;    //    65,536
    const size_t NE = (size_t)NC * EC;        // 4,194,304
    const size_t NN = (size_t)NC * NC;        // 16,777,216

    // ---- f32 scratch ----
    float* f = (float*)d_ws;
    float* hhe    = f; f += ND;
    float* hn     = f; f += ND;
    float* hg     = f; f += ND;
    float* sc     = f; f += NC;
    float* s1     = f; f += NC;
    float* s2     = f; f += NC;
    float* colmax = f; f += EC;
    float* colsum = f; f += EC;
    float* hesz   = f; f += EC;
    float* Am     = f; f += NE;
    float* Bm     = f; f += NE;
    float* Mm     = f; f += NE;
    float* Tm     = f; f += NE;   // A_hat^T, then B_hat
    float* Gm     = f; f += NN;
    // ---- f16 mirrors (region starts 16B-aligned; all sizes %8==0) ----
    _Float16* h = (_Float16*)f;
    _Float16* x16      = h; h += ND;
    _Float16* Whe16    = h; h += DD;
    _Float16* We16     = h; h += DD;
    _Float16* Wn16     = h; h += DD;
    _Float16* Wg16     = h; h += DD;
    _Float16* H16      = h; h += NE;
    _Float16* hhe16    = h; h += ND;
    _Float16* hn16     = h; h += ND;
    _Float16* hg16     = h; h += ND;
    _Float16* heW16    = h; h += ED;
    _Float16* emix16   = h; h += ED;
    _Float16* emixWe16 = h; h += ED;
    _Float16* Am16     = h; h += NE;
    _Float16* Bm16     = h; h += NE;
    _Float16* Tm16     = h; h += NE;
    _Float16* Gm16     = h; h += NN;

    const dim3 blk(256);
    auto gg = [](int M, int N) { return dim3(N / TILE_N, M / TILE_M); };

    // ---- one-time f16 conversions of raw inputs ----
    cvt_f16_kernel<<<(unsigned)(ND / 256), blk, 0, stream>>>(x, x16, ND);
    cvt_f16_kernel<<<(unsigned)(DD / 256), blk, 0, stream>>>(W_he, Whe16, DD);
    cvt_f16_kernel<<<(unsigned)(DD / 256), blk, 0, stream>>>(W_e,  We16,  DD);
    cvt_f16_kernel<<<(unsigned)(DD / 256), blk, 0, stream>>>(W_n,  Wn16,  DD);
    cvt_f16_kernel<<<(unsigned)(DD / 256), blk, 0, stream>>>(W_g,  Wg16,  DD);
    cvt_f16_kernel<<<(unsigned)(NE / 256), blk, 0, stream>>>(H, H16, NE);

    // h_he = x@W_he ; hn = x@W_n ; h_g = x@W_g   (dual f32+f16 outputs)
    gemm_wmma_f16<<<gg(NC, DIMC), blk, 0, stream>>>(x16, DIMC, 0, Whe16, DIMC, 0, hhe, DIMC, hhe16, nullptr, nullptr, 1.f, NC, DIMC, DIMC, 0);
    gemm_wmma_f16<<<gg(NC, DIMC), blk, 0, stream>>>(x16, DIMC, 0, Wn16,  DIMC, 0, hn,  DIMC, hn16,  nullptr, nullptr, 1.f, NC, DIMC, DIMC, 0);
    gemm_wmma_f16<<<gg(NC, DIMC), blk, 0, stream>>>(x16, DIMC, 0, Wg16,  DIMC, 0, hg,  DIMC, hg16,  nullptr, nullptr, 1.f, NC, DIMC, DIMC, 0);

    // sc, s1, s2
    dots_kernel<<<NC, blk, 0, stream>>>(hhe, hg, u, a_g, sc, s1, s2, INV_SQRT_D);

    // A (column softmax over masked sc)
    col_stats_kernel<<<EC / 256, blk, 0, stream>>>(H, sc, colmax, colsum, hesz);
    build_A_kernel<<<(unsigned)(NE / 256), blk, 0, stream>>>(H, sc, colmax, colsum, Am, Am16);

    // e_repr = A^T @ h_he ; he = e_repr @ W_e (f16 only)
    gemm_wmma_f16<<<gg(EC, DIMC), blk, 0, stream>>>(Am16, EC, 1, hhe16, DIMC, 0, nullptr, DIMC, emix16, nullptr, nullptr, 1.f, EC, DIMC, NC, 0);
    gemm_wmma_f16<<<gg(EC, DIMC), blk, 0, stream>>>(emix16, DIMC, 0, We16, DIMC, 0, nullptr, DIMC, heW16, nullptr, nullptr, 1.f, EC, DIMC, DIMC, 0);

    // B = rowsoftmax_mask( (hn @ he^T) * inv )
    gemm_wmma_f16<<<gg(NC, EC), blk, 0, stream>>>(hn16, DIMC, 0, heW16, DIMC, 1, Bm, EC, nullptr, nullptr, nullptr, INV_SQRT_D, NC, EC, DIMC, 0);
    row_softmax_kernel<<<NC, blk, 0, stream>>>(Bm, H, EC, 1, 1, Bm16);

    // G = rowsoftmax( masked lrelu(s1_i + s2_j) )
    build_G_kernel<<<NC, blk, 0, stream>>>(adj, s1, s2, Gm);
    row_softmax_kernel<<<NC, blk, 0, stream>>>(Gm, nullptr, NC, 0, 0, Gm16);

    // M = (G @ H) / he_sz
    gemm_wmma_f16<<<gg(NC, EC), blk, 0, stream>>>(Gm16, NC, 0, H16, EC, 0, Mm, EC, nullptr, nullptr, hesz, 1.f, NC, EC, NC, 0);

    // A_hat^T = colsoftmax(A + M) * mask ; e_mix = A_hat @ h_he ; e_mix @ W_e
    col_softmax_kernel<<<EC / 256, blk, 0, stream>>>(Am, Mm, H, Tm, Tm16);
    gemm_wmma_f16<<<gg(EC, DIMC), blk, 0, stream>>>(Tm16, EC, 1, hhe16, DIMC, 0, nullptr, DIMC, emix16, nullptr, nullptr, 1.f, EC, DIMC, NC, 0);
    gemm_wmma_f16<<<gg(EC, DIMC), blk, 0, stream>>>(emix16, DIMC, 0, We16, DIMC, 0, nullptr, DIMC, emixWe16, nullptr, nullptr, 1.f, EC, DIMC, DIMC, 0);

    // B_hat = rowsoftmax(B + M) * mask ; z_h = elu(B_hat @ (e_mix@W_e))
    add_kernel<<<(unsigned)(NE / 256), blk, 0, stream>>>(Bm, Mm, Tm, NE);
    row_softmax_kernel<<<NC, blk, 0, stream>>>(Tm, H, EC, 0, 1, Tm16);
    gemm_wmma_f16<<<gg(NC, DIMC), blk, 0, stream>>>(Tm16, EC, 0, emixWe16, DIMC, 0, out, DIMC, nullptr, nullptr, nullptr, 1.f, NC, DIMC, EC, 1);

    // G_hat = rowsoftmax(G + B@B^T + A@A^T)  (accumulated in place into G)
    gemm_wmma_f16<<<gg(NC, NC), blk, 0, stream>>>(Bm16, EC, 0, Bm16, EC, 1, Gm, NC, nullptr, Gm, nullptr, 1.f, NC, NC, EC, 0);
    gemm_wmma_f16<<<gg(NC, NC), blk, 0, stream>>>(Am16, EC, 0, Am16, EC, 1, Gm, NC, nullptr, Gm, nullptr, 1.f, NC, NC, EC, 0);
    row_softmax_kernel<<<NC, blk, 0, stream>>>(Gm, nullptr, NC, 0, 0, Gm16);

    // z_g = elu(G_hat @ h_g)
    gemm_wmma_f16<<<gg(NC, DIMC), blk, 0, stream>>>(Gm16, NC, 0, hg16, DIMC, 0, out + ND, DIMC, nullptr, nullptr, nullptr, 1.f, NC, DIMC, NC, 1);
}
